// LBP_block_62783831933702
// MI455X (gfx1250) — matrix operand order repaired and checked by reading, not compiled, not run on hardware
//
#include <hip/hip_runtime.h>
#include <stdint.h>

typedef __attribute__((ext_vector_type(16))) _Float16 v16h;
typedef __attribute__((ext_vector_type(8)))  float    v8f;
typedef __attribute__((ext_vector_type(4)))  float    v4f;
typedef __attribute__((ext_vector_type(2)))  _Float16 h2;
typedef __attribute__((ext_vector_type(4)))  unsigned int v4u;
typedef __attribute__((ext_vector_type(8)))  int      v8i;
typedef __attribute__((ext_vector_type(4)))  int      v4i;

#define H_IMG 224
#define W_IMG 224
#define NPIX  64      // pixels per workgroup tile (4 WMMA N-tiles)
#define LPAD  72      // padded channel stride in LDS (144B rows -> conflict-free)
#define WROWS 640     // 576 ternary-weight rows + 64 w1x1 rows

// ---------------------------------------------------------------------------
// Kernel 0: one-shot weight pre-pack to f16 in TDM-friendly layout:
//   rows 0..575  : wall[t*64+co][ci] = w_bin[co][ci][t]   (tap-major)
//   rows 576..639: wall[576+co][ci] = w1x1[co][ci]
// Contiguous 640 x 64 halfs (80KB) in d_ws.
// ---------------------------------------------------------------------------
__global__ __launch_bounds__(256) void lbp_pack_weights(
    const float* __restrict__ wb, const float* __restrict__ w1,
    _Float16* __restrict__ wf16)
{
    int o = blockIdx.x * 256 + threadIdx.x;     // 0 .. 640*64-1
    if (o >= WROWS * 64) return;
    int row = o >> 6, ci = o & 63;
    float v;
    if (row < 576) {
        int t = row >> 6, co = row & 63;
        v = wb[co * 576 + ci * 9 + t];
    } else {
        v = w1[(row - 576) * 64 + ci];
    }
    wf16[o] = (_Float16)v;
}

// ---------------------------------------------------------------------------
// Kernel 1: fused ternary 3x3 conv + 1x1 conv + bias, writes y to out and
// accumulates per-channel sum / sumsq into stats[0..127].
// Weights arrive in LDS via TDM (tensor_load_to_lds) with hardware row
// padding 64->72 halfs; x slab is staged by the VALU concurrently.
// Grid: (4 w-tiles, 224 h, 8 b), block = 128 threads (4 waves).
// ---------------------------------------------------------------------------
__global__ __launch_bounds__(128) void lbp_fused_conv(
    const float* __restrict__ x,        // [8][64][224][224]
    const _Float16* __restrict__ wf16,  // [640][64] prepacked weights
    const float* __restrict__ b1,       // [64]
    float* __restrict__ out,            // [8][64][224][224] (pre-BN y)
    float* __restrict__ stats)          // [0..63]=sum, [64..127]=sumsq
{
    __shared__ _Float16 wall[WROWS][LPAD];  // rows 0..575 taps, 576..639 w1x1
    __shared__ _Float16 xsh[3][66][LPAD];   // [row][col][ci]  ci fastest
    __shared__ _Float16 tsh[NPIX][LPAD];    // [pixel][ci]  (t transposed)

    const int tid  = threadIdx.x;
    const int lane = tid & 31;
    const int wv   = tid >> 5;          // wave id 0..3
    const int w0   = blockIdx.x * NPIX; // pixel tile start
    const int h    = blockIdx.y;
    const int b    = blockIdx.z;

    // ---- TDM: DMA 160 weight rows per wave, global->LDS, pad 64->72 halfs --
    {
        const unsigned lds_base =
            (unsigned)(uintptr_t)(void*)&wall[0][0] + (unsigned)(wv * 160 * LPAD * 2);
        const unsigned long long ga =
            (unsigned long long)(uintptr_t)wf16 + (unsigned long long)(wv * 160 * 128);
        v4u g0;
        g0[0] = 1u;                                         // count=1 (valid)
        g0[1] = lds_base;                                   // lds_addr
        g0[2] = (unsigned)(ga & 0xffffffffu);               // global_addr lo
        g0[3] = (unsigned)((ga >> 32) & 0x1ffffffu)         // global_addr hi
              | (2u << 30);                                 // type = 2 (image)
        v8i g1 = {};
        g1[0] = (3 << 16)     // data_size = 8B units
              | (1 << 20)     // pad_enable
              | (4 << 22)     // pad_interval: 32 DWORDs (one 128B row)
              | (3 << 25);    // pad_amount: 4 DWORDs (8 halfs)
        g1[1] = 16 << 16;     // tensor_dim0 = 16 x 8B = 128B row
        g1[2] = 160 << 16;    // tensor_dim1 = 160 rows (this wave's slice)
        g1[3] = 16 << 16;     // tile_dim0   = 16
        g1[4] = 160;          // tile_dim1   = 160
        g1[5] = 16;           // tensor_dim0_stride = 16 x 8B
        v4i g2 = {};
        v4i g3 = {};
        v8i g4 = {};
        __builtin_amdgcn_tensor_load_to_lds(g0, g1, g2, g3, g4, 0);
    }

    // ---- stage x slab (3 rows x 66 cols x 64 ch) f32->f16, runs under DMA --
    for (int g = tid; g < 64 * 3 * 66; g += 128) {
        int ci = g / 198;
        int rr = g - ci * 198;
        int r  = rr / 66;
        int cc = rr - r * 66;
        int hs = h - 1 + r;
        int ws = w0 - 1 + cc;
        float v = 0.0f;
        if ((unsigned)hs < H_IMG && (unsigned)ws < W_IMG)
            v = x[(((b * 64 + ci) * H_IMG) + hs) * W_IMG + ws];
        xsh[r][cc][ci] = (_Float16)v;
    }
    __builtin_amdgcn_s_wait_tensorcnt(0);
    __syncthreads();

    const int nlo  = lane & 15;   // N column / M row within tile
    const int hsel = lane >> 4;   // half-wave select
    const int co   = (wv << 4) + nlo;

    // ======================= stage 1: 3x3 ternary conv =====================
    v8f acc[4] = {};
    for (int t = 0; t < 9; ++t) {
        const int ky = t / 3, kx = t - ky * 3;
        for (int kc = 0; kc < 2; ++kc) {
            // A fragment (16x32): lanes 0-15: K 0-15, lanes 16-31: K 16-31
            v16h a;
            #pragma unroll
            for (int v = 0; v < 8; ++v) {
                int kl = (hsel << 4) + 2 * v;
                h2 p = *(const h2*)&wall[t * 64 + co][kc * 32 + kl];
                a[2 * v] = p[0]; a[2 * v + 1] = p[1];
            }
            #pragma unroll
            for (int nt = 0; nt < 4; ++nt) {
                // B fragment (32x16): lane n, half-wave splits K 0-15/16-31
                v16h bm;
                const int cc = nt * 16 + nlo + kx;
                #pragma unroll
                for (int v = 0; v < 8; ++v) {
                    int ci = kc * 32 + (hsel << 4) + 2 * v;
                    h2 p = *(const h2*)&xsh[ky][cc][ci];
                    bm[2 * v] = p[0]; bm[2 * v + 1] = p[1];
                }
                acc[nt] = __builtin_amdgcn_wmma_f32_16x16x32_f16(
                    false, a, false, bm, (short)0, acc[nt], false, false);
            }
        }
    }

    // ---- spill t to LDS transposed: tsh[n][c], f16 ------------------------
    #pragma unroll
    for (int nt = 0; nt < 4; ++nt)
        #pragma unroll
        for (int v = 0; v < 8; ++v)
            tsh[nt * 16 + nlo][(wv << 4) + v + (hsel << 3)] =
                (_Float16)acc[nt][v];
    __syncthreads();

    // ======================= stage 2: 1x1 conv =============================
    v8f acc2[4] = {};
    for (int kc = 0; kc < 2; ++kc) {
        v16h a;
        #pragma unroll
        for (int v = 0; v < 8; ++v) {
            int kl = (hsel << 4) + 2 * v;
            h2 p = *(const h2*)&wall[576 + co][kc * 32 + kl];
            a[2 * v] = p[0]; a[2 * v + 1] = p[1];
        }
        #pragma unroll
        for (int nt = 0; nt < 4; ++nt) {
            v16h bm;
            #pragma unroll
            for (int v = 0; v < 8; ++v) {
                int k = kc * 32 + (hsel << 4) + 2 * v;
                h2 p = *(const h2*)&tsh[nt * 16 + nlo][k];
                bm[2 * v] = p[0]; bm[2 * v + 1] = p[1];
            }
            acc2[nt] = __builtin_amdgcn_wmma_f32_16x16x32_f16(
                false, a, false, bm, (short)0, acc2[nt], false, false);
        }
    }

    // ======================= bias, store y, channel stats ==================
    float bias[8];
    #pragma unroll
    for (int v = 0; v < 8; ++v)
        bias[v] = b1[(wv << 4) + v + (hsel << 3)];

    float ls[8], ls2[8];
    #pragma unroll
    for (int v = 0; v < 8; ++v) { ls[v] = 0.0f; ls2[v] = 0.0f; }

    #pragma unroll
    for (int nt = 0; nt < 4; ++nt) {
        const int n  = nt * 16 + nlo;
        const int wp = w0 + n;
        const bool ok = wp < W_IMG;
        #pragma unroll
        for (int v = 0; v < 8; ++v) {
            const int ch = (wv << 4) + v + (hsel << 3);
            float val = acc2[nt][v] + bias[v];
            if (ok) {
                out[(((b * 64 + ch) * H_IMG) + h) * W_IMG + wp] = val;
                ls[v]  += val;
                ls2[v] += val * val;
            }
        }
    }
    // reduce across the 16 lanes of each half-wave, then 2 atomics per (wv,v)
    #pragma unroll
    for (int v = 0; v < 8; ++v) {
        float s = ls[v], s2 = ls2[v];
        #pragma unroll
        for (int off = 8; off >= 1; off >>= 1) {
            s  += __shfl_xor(s,  off, 32);
            s2 += __shfl_xor(s2, off, 32);
        }
        if (nlo == 0) {
            const int ch = (wv << 4) + v + (hsel << 3);
            atomicAdd(&stats[ch],      s);
            atomicAdd(&stats[64 + ch], s2);
        }
    }
}

// ---------------------------------------------------------------------------
// Kernel 2: fold batch stats into per-channel scale/shift.
// ---------------------------------------------------------------------------
__global__ void lbp_stats_finalize(const float* __restrict__ gamma,
                                   const float* __restrict__ beta,
                                   float* __restrict__ stats)
{
    int c = threadIdx.x;
    if (c < 64) {
        const float N = 8.0f * 224.0f * 224.0f;
        float mean = stats[c] / N;
        float var  = stats[64 + c] / N - mean * mean;
        float sc   = gamma[c] * rsqrtf(var + 1e-5f);
        stats[128 + c] = sc;
        stats[192 + c] = beta[c] - mean * sc;
    }
}

// ---------------------------------------------------------------------------
// Kernel 3: in-place BN + leaky-ReLU(0.1) + clamp[0,255], v4f streaming.
// Loads hit L2 (y was just produced); final store is nontemporal.
// ---------------------------------------------------------------------------
__global__ __launch_bounds__(256) void lbp_bn_act(float* __restrict__ out,
                                                  const float* __restrict__ stats)
{
    const int i = blockIdx.x * 256 + threadIdx.x;       // v4f index
    const int c = (i / 12544) & 63;                     // 50176/4 per plane
    const float sc = stats[128 + c];
    const float sh = stats[192 + c];
    v4f y = *((const v4f*)out + i);
    #pragma unroll
    for (int j = 0; j < 4; ++j) {
        float v = y[j] * sc + sh;
        v = (v >= 0.0f) ? v : 0.1f * v;
        v = fminf(fmaxf(v, 0.0f), 255.0f);
        y[j] = v;
    }
    __builtin_nontemporal_store(y, (v4f*)out + i);
}

// ---------------------------------------------------------------------------
extern "C" void kernel_launch(void* const* d_in, const int* in_sizes, int n_in,
                              void* d_out, int out_size, void* d_ws, size_t ws_size,
                              hipStream_t stream)
{
    const float* x     = (const float*)d_in[0];
    const float* w_bin = (const float*)d_in[1];
    const float* w1x1  = (const float*)d_in[2];
    const float* b1x1  = (const float*)d_in[3];
    const float* gamma = (const float*)d_in[4];
    const float* beta  = (const float*)d_in[5];
    float* out   = (float*)d_out;
    float* stats = (float*)d_ws;
    _Float16* wf16 = (_Float16*)((char*)d_ws + 1024);   // 80KB weight pack

    (void)hipMemsetAsync(stats, 0, 256 * sizeof(float), stream);

    lbp_pack_weights<<<(WROWS * 64) / 256, 256, 0, stream>>>(w_bin, w1x1, wf16);

    dim3 grid(4, H_IMG, 8);          // 4 w-tiles x 224 rows x 8 batch
    lbp_fused_conv<<<grid, 128, 0, stream>>>(x, wf16, b1x1, out, stats);
    lbp_stats_finalize<<<1, 64, 0, stream>>>(gamma, beta, stats);

    const int n4 = (8 * 64 * H_IMG * W_IMG) / 4;   // 6,422,528 float4
    lbp_bn_act<<<n4 / 256, 256, 0, stream>>>(out, stats);
}
